// Skeleton4D3DBlock_24180665876978
// MI455X (gfx1250) — matrix operand order repaired and checked by reading, not compiled, not run on hardware
//
#include <hip/hip_runtime.h>
#include <hip/hip_bf16.h>

// ---------------------------------------------------------------------------
// Model dims (fixed by the reference)
// ---------------------------------------------------------------------------
#define BSZ     2
#define TSEQ    2048
#define DM      1024
#define NH      16
#define DH      64
#define DFF     4096
#define NFAM    64
#define NDICT   16384
#define NTOK    (BSZ*TSEQ)          // 4096
#define TOPK    8
#define EPSV    1e-6f

typedef unsigned short u16;
typedef unsigned long long uptr;
typedef int i32x4 __attribute__((vector_size(16)));   // builtin's pointee type
typedef __attribute__((ext_vector_type(16))) __bf16 v16bf;
typedef __attribute__((ext_vector_type(8)))  float  v8f;

// ---------------------------------------------------------------------------
// CDNA5 async global->LDS copy (ASYNCcnt-tracked DMA), with sync fallback so
// the source compiles on toolchains lacking the builtin. Builtin signature
// (from the compiler diagnostic): (int4 AS1*, int4 AS3*, imm offset, imm cpol).
// Generic->AS casts go through an integer: flat LDS pointers carry the LDS
// offset in the low 32 bits (ISA 10.2 aperture rules).
// ---------------------------------------------------------------------------
#if defined(__gfx1250__) && __has_builtin(__builtin_amdgcn_global_load_async_to_lds_b128)
#define GLD_ASYNC_B128(gp, lp)                                               \
    __builtin_amdgcn_global_load_async_to_lds_b128(                          \
        (__attribute__((address_space(1))) i32x4*)(uptr)(gp),                \
        (__attribute__((address_space(3))) i32x4*)(uptr)(lp), 0, 0)
#else
#define GLD_ASYNC_B128(gp, lp) (*(uint4*)(lp) = *(const uint4*)(gp))
#endif

#if defined(__gfx1250__) && __has_builtin(__builtin_amdgcn_s_wait_asynccnt)
#define WAIT_ASYNC() __builtin_amdgcn_s_wait_asynccnt(0)
#elif defined(__gfx1250__)
#define WAIT_ASYNC() asm volatile("s_wait_asynccnt 0x0" ::: "memory")
#else
#define WAIT_ASYNC() ((void)0)
#endif

__device__ inline u16 f2bf(float f) {
    unsigned u = __builtin_bit_cast(unsigned, f);
    u += 0x7fffu + ((u >> 16) & 1u);          // round-to-nearest-even
    return (u16)(u >> 16);
}

union FragBF { uint4 q[2]; v16bf v; };        // 32B fragment (16 bf16)

// ---------------------------------------------------------------------------
// Batched strided WMMA GEMM:  C[z][M,N] = alpha * A[z][M,K] * B[z][N,K]^T
//   A: f32 (converted to bf16 while staging, sync) or bf16 (async DMA to LDS)
//   B: always bf16 (async DMA to LDS)
//   C: f32 or bf16 (OBF)
// Requirements: M%128==0, N%64==0, K%32==0 (all call sites satisfy this).
// Block = 256 thr = 8 waves; block tile 128x64; wave tile 32x32 (2x2 WMMA).
// Double-buffered LDS: async loads of chunk i+1 overlap WMMAs on chunk i.
// Per-z base offset = (z/zsplit)*s?1 + (z%zsplit)*s?2   (elements).
// ---------------------------------------------------------------------------
template<bool ABF, bool OBF>
__global__ __launch_bounds__(256) void gemm_nt_k(
    const void* __restrict__ A_, const u16* __restrict__ B_, void* __restrict__ C_,
    int K, int lda, int ldb, int ldc,
    long long sA1, long long sA2, long long sB1, long long sB2,
    long long sC1, long long sC2, int zsplit, float alpha)
{
    constexpr int LD = 40;                     // padded LDS row stride (halves)
    __shared__ u16 lA[2][128 * LD];
    __shared__ u16 lB[2][64  * LD];

    const int tid = threadIdx.x;
    const int z   = blockIdx.z;
    const int z1  = z / zsplit, z2 = z - z1 * zsplit;

    const float* Af = (const float*)A_;
    const u16*   Ah = (const u16*)A_;
    const size_t arow0 = (size_t)z1 * (size_t)sA1 + (size_t)z2 * (size_t)sA2
                       + (size_t)(blockIdx.y * 128) * (size_t)lda;
    const u16* Bp = B_ + (size_t)z1 * (size_t)sB1 + (size_t)z2 * (size_t)sB2
                       + (size_t)(blockIdx.x * 64) * (size_t)ldb;

    const int lane  = tid & 31;
    const int wave  = tid >> 5;
    const int waveM = (wave >> 1) * 32;        // 0..96
    const int waveN = (wave & 1)  * 32;        // 0/32
    const int l15   = lane & 15;
    const int hi    = lane >> 4;               // half-wave select
    const int akb   = hi * 8;                  // A frag K base (ISA 7.12.2)
    const int bkb   = hi * 8;                  // B frag K base

    // stage tiles for K-chunk starting at kb into buffer bufi
    auto stage = [&](int kb, int bufi) {
        // B tile 64x32 bf16: 256 x 16B DMA chunks, one per thread
        {
            int row = tid >> 2, c8 = (tid & 3) * 8;
            GLD_ASYNC_B128(Bp + (size_t)row * ldb + kb + c8,
                           &lB[bufi][row * LD + c8]);
        }
        if constexpr (ABF) {
            // A tile 128x32 bf16: 512 x 16B DMA chunks, two per thread
            #pragma unroll
            for (int i = 0; i < 2; ++i) {
                int g = tid + i * 256;
                int row = g >> 2, c8 = (g & 3) * 8;
                GLD_ASYNC_B128(Ah + arow0 + (size_t)row * lda + kb + c8,
                               &lA[bufi][row * LD + c8]);
            }
        } else {
            // A tile 128x32 from f32: convert in flight (sync ds_store path)
            #pragma unroll
            for (int i = 0; i < 4; ++i) {
                int g = tid + i * 256;
                int row = g >> 3, c4 = (g & 7) * 4;
                float4 f = *(const float4*)(Af + arow0 + (size_t)row * lda + kb + c4);
                uint2 p;
                p.x = (unsigned)f2bf(f.x) | ((unsigned)f2bf(f.y) << 16);
                p.y = (unsigned)f2bf(f.z) | ((unsigned)f2bf(f.w) << 16);
                *(uint2*)&lA[bufi][row * LD + c4] = p;
            }
        }
    };

    v8f acc[2][2];
    #pragma unroll
    for (int i = 0; i < 2; ++i)
        #pragma unroll
        for (int j = 0; j < 2; ++j) {
            v8f zz = {0.f,0.f,0.f,0.f,0.f,0.f,0.f,0.f};
            acc[i][j] = zz;
        }

    const int nk = K >> 5;
    stage(0, 0);
    for (int ik = 0; ik < nk; ++ik) {
        WAIT_ASYNC();                 // this wave's DMA for chunk ik done
        __syncthreads();              // ... and every other wave's too
        if (ik + 1 < nk) stage((ik + 1) << 5, (ik + 1) & 1);

        const u16* la = &lA[ik & 1][0];
        const u16* lb = &lB[ik & 1][0];
        FragBF a[2], b[2];
        #pragma unroll
        for (int t = 0; t < 2; ++t) {
            const u16* ra = &la[(waveM + t*16 + l15) * LD];
            a[t].q[0] = *(const uint4*)(ra + akb);          // K = akb..akb+7
            a[t].q[1] = *(const uint4*)(ra + akb + 16);     // K = akb+16..akb+23
            const u16* rb = &lb[(waveN + t*16 + l15) * LD];
            b[t].q[0] = *(const uint4*)(rb + bkb);          // K = bkb..bkb+7
            b[t].q[1] = *(const uint4*)(rb + bkb + 16);     // K = bkb+16..bkb+23
        }
        #pragma unroll
        for (int tm = 0; tm < 2; ++tm)
            #pragma unroll
            for (int tn = 0; tn < 2; ++tn)
                acc[tm][tn] = __builtin_amdgcn_wmma_f32_16x16x32_bf16(
                    false, a[tm].v, false, b[tn].v,
                    (short)0, acc[tm][tn], false, false);
    }

    // ---- store: VGPR v -> row v (+8 for hi half-wave) ----
    const size_t coff = (size_t)z1 * (size_t)sC1 + (size_t)z2 * (size_t)sC2;
    const int mBase = blockIdx.y * 128 + waveM + hi * 8;
    const int nBase = blockIdx.x * 64  + waveN + l15;
    #pragma unroll
    for (int tm = 0; tm < 2; ++tm)
        #pragma unroll
        for (int tn = 0; tn < 2; ++tn)
            #pragma unroll
            for (int v = 0; v < 8; ++v) {
                size_t idx = coff + (size_t)(mBase + tm*16 + v) * ldc + (nBase + tn*16);
                float val = acc[tm][tn][v] * alpha;
                if constexpr (OBF) ((u16*)C_)[idx] = f2bf(val);
                else               ((float*)C_)[idx] = val;
            }
}

// f32 -> bf16 stream conversion (4 elems/thread); n % 1024 == 0
__global__ void f32_to_bf16_k(const float* __restrict__ in, u16* __restrict__ out)
{
    size_t i4 = ((size_t)blockIdx.x * 256 + threadIdx.x) * 4;
    float4 f = *(const float4*)(in + i4);
    uint2 p;
    p.x = (unsigned)f2bf(f.x) | ((unsigned)f2bf(f.y) << 16);
    p.y = (unsigned)f2bf(f.z) | ((unsigned)f2bf(f.w) << 16);
    *(uint2*)(out + i4) = p;
}

// ---------------------------------------------------------------------------
// Family softmax over 64: one wave per token (wave32: 2 elems/lane)
// ---------------------------------------------------------------------------
__global__ void softmax64_k(const float* __restrict__ in, float* __restrict__ out)
{
    int wave = threadIdx.x >> 5, lane = threadIdx.x & 31;
    int token = blockIdx.x * 8 + wave;
    const float* r = in + (size_t)token * NFAM;
    float a0 = r[lane], a1 = r[lane + 32];
    float m = fmaxf(a0, a1);
    #pragma unroll
    for (int o = 16; o > 0; o >>= 1) m = fmaxf(m, __shfl_xor(m, o, 32));
    float e0 = __expf(a0 - m), e1 = __expf(a1 - m);
    float s = e0 + e1;
    #pragma unroll
    for (int o = 16; o > 0; o >>= 1) s += __shfl_xor(s, o, 32);
    float inv = 1.f / s;
    out[(size_t)token * NFAM + lane]      = e0 * inv;
    out[(size_t)token * NFAM + lane + 32] = e1 * inv;
}

// basis[t,d] = sum_k fam[t,k] * proto[k,d]   (K=64, tiny -> VALU)
__global__ void basis_k(const float* __restrict__ fam, const float* __restrict__ proto,
                        float* __restrict__ basis)
{
    __shared__ float s[NFAM];
    int token = blockIdx.x;
    if (threadIdx.x < NFAM) s[threadIdx.x] = fam[(size_t)token * NFAM + threadIdx.x];
    __syncthreads();
    int d0 = threadIdx.x * 4;
    float a0 = 0, a1 = 0, a2 = 0, a3 = 0;
    for (int k = 0; k < NFAM; ++k) {
        const float* p = proto + (size_t)k * DM + d0;
        float sv = s[k];
        a0 += sv * p[0]; a1 += sv * p[1]; a2 += sv * p[2]; a3 += sv * p[3];
    }
    float* o = basis + (size_t)token * DM + d0;
    o[0] = a0; o[1] = a1; o[2] = a2; o[3] = a3;
}

__global__ void sub_k(const float* a, const float* b, float* c) {
    size_t i = (size_t)blockIdx.x * 256 + threadIdx.x;
    c[i] = a[i] - b[i];
}
__global__ void add_k(const float* a, const float* b, float* c) {
    size_t i = (size_t)blockIdx.x * 256 + threadIdx.x;
    c[i] = a[i] + b[i];
}
__global__ void add_inplace_k(float* c, const float* a) {
    size_t i = (size_t)blockIdx.x * 256 + threadIdx.x;
    c[i] += a[i];
}

// ---------------------------------------------------------------------------
// Iterative deterministic top-8 over 16384 coeffs (one block per token)
// ---------------------------------------------------------------------------
__global__ void topk8_k(const float* __restrict__ coeffs,
                        float* __restrict__ tv, int* __restrict__ ti, int tok_base)
{
    const float* row = coeffs + (size_t)blockIdx.x * NDICT;
    __shared__ float bv[256]; __shared__ int bi[256];
    __shared__ float cv[TOPK]; __shared__ int ci[TOPK];
    int tid = threadIdx.x;
    for (int sel = 0; sel < TOPK; ++sel) {
        float best = -3.4e38f; int bidx = 1 << 30;
        for (int i = tid; i < NDICT; i += 256) {
            bool skip = false;
            for (int c = 0; c < sel; ++c) if (ci[c] == i) skip = true;
            if (skip) continue;
            float v = row[i];
            if (v > best || (v == best && i < bidx)) { best = v; bidx = i; }
        }
        bv[tid] = best; bi[tid] = bidx; __syncthreads();
        for (int s = 128; s > 0; s >>= 1) {
            if (tid < s) {
                float ov = bv[tid + s]; int oi = bi[tid + s];
                if (ov > bv[tid] || (ov == bv[tid] && oi < bi[tid])) { bv[tid] = ov; bi[tid] = oi; }
            }
            __syncthreads();
        }
        if (tid == 0) { cv[sel] = bv[0]; ci[sel] = bi[0]; }
        __syncthreads();
    }
    if (tid < TOPK) {
        int t = tok_base + blockIdx.x;
        tv[(size_t)t * TOPK + tid] = cv[tid];
        ti[(size_t)t * TOPK + tid] = ci[tid];
    }
}

// ---------------------------------------------------------------------------
// Fused x_rec = basis + sum_k val*dict[idx] + bias, then RMSNorm -> bf16
// ---------------------------------------------------------------------------
__global__ void sparse_norm_k(const float* __restrict__ tv, const int* __restrict__ ti,
                              const float* __restrict__ basis, const float* __restrict__ dictw,
                              const float* __restrict__ bias, const float* __restrict__ w,
                              u16* __restrict__ normed, float* __restrict__ rms_out)
{
    __shared__ float sv[TOPK]; __shared__ int si[TOPK]; __shared__ float red[256];
    int token = blockIdx.x, tid = threadIdx.x;
    if (tid < TOPK) { sv[tid] = tv[(size_t)token*TOPK + tid]; si[tid] = ti[(size_t)token*TOPK + tid]; }
    __syncthreads();
    int d0 = tid * 4;
    const float* b = basis + (size_t)token * DM + d0;
    float xr[4];
    #pragma unroll
    for (int j = 0; j < 4; ++j) xr[j] = b[j] + bias[d0 + j];
    for (int k = 0; k < TOPK; ++k) {
        const float* dr = dictw + (size_t)si[k] * DM + d0;
        float c = sv[k];
        #pragma unroll
        for (int j = 0; j < 4; ++j) xr[j] += c * dr[j];
    }
    float ss = xr[0]*xr[0] + xr[1]*xr[1] + xr[2]*xr[2] + xr[3]*xr[3];
    red[tid] = ss; __syncthreads();
    for (int s = 128; s > 0; s >>= 1) { if (tid < s) red[tid] += red[tid + s]; __syncthreads(); }
    float rms = sqrtf(red[0] * (1.f / DM) + EPSV);
    if (tid == 0) rms_out[token] = rms;
    float inv = 1.f / rms;
    u16* o = normed + (size_t)token * DM + d0;
    #pragma unroll
    for (int j = 0; j < 4; ++j) o[j] = f2bf(w[d0 + j] * xr[j] * inv);
}

// Plain RMSNorm -> bf16 (for the second norm)
__global__ void rmsnorm_k(const float* __restrict__ x, const float* __restrict__ w,
                          u16* __restrict__ normed, float* __restrict__ rms_out)
{
    __shared__ float red[256];
    int token = blockIdx.x, tid = threadIdx.x, d0 = tid * 4;
    const float* r = x + (size_t)token * DM + d0;
    float xr[4] = {r[0], r[1], r[2], r[3]};
    red[tid] = xr[0]*xr[0] + xr[1]*xr[1] + xr[2]*xr[2] + xr[3]*xr[3];
    __syncthreads();
    for (int s = 128; s > 0; s >>= 1) { if (tid < s) red[tid] += red[tid + s]; __syncthreads(); }
    float rms = sqrtf(red[0] * (1.f / DM) + EPSV);
    if (tid == 0) rms_out[token] = rms;
    float inv = 1.f / rms;
    u16* o = normed + (size_t)token * DM + d0;
    #pragma unroll
    for (int j = 0; j < 4; ++j) o[j] = f2bf(w[d0 + j] * xr[j] * inv);
}

// vT[b,h,e,t] = bf16(vlin[b,t,h*64+e])
__global__ void transpose_vT_k(const float* __restrict__ vlin, u16* __restrict__ vT)
{
    size_t o = (size_t)blockIdx.x * 256 + threadIdx.x;
    int t = (int)(o & 2047), e = (int)((o >> 11) & 63), h = (int)((o >> 17) & 15), b = (int)(o >> 21);
    vT[o] = f2bf(vlin[((size_t)(b * TSEQ + t)) * DM + h * DH + e]);
}
// relT[h,e,d] = bf16(rel[h,d,e])
__global__ void transpose_rel_k(const float* __restrict__ rel, u16* __restrict__ relT)
{
    size_t o = (size_t)blockIdx.x * 256 + threadIdx.x;
    int d = (int)(o & 63), e = (int)((o >> 6) & 63), h = (int)(o >> 12);
    relT[(size_t)h * 4096 + e * 64 + d] = f2bf(rel[(size_t)h * 4096 + d * 64 + e]);
}

// In-place row softmax over 2048 (attention probabilities, kept f32 in d_out)
__global__ void attn_softmax_k(float* __restrict__ attn)
{
    __shared__ float red[256];
    float* row = attn + (size_t)blockIdx.x * TSEQ;
    int tid = threadIdx.x;
    float v[8]; float m = -3.4e38f;
    #pragma unroll
    for (int j = 0; j < 8; ++j) { v[j] = row[tid + j*256]; m = fmaxf(m, v[j]); }
    red[tid] = m; __syncthreads();
    for (int s = 128; s > 0; s >>= 1) { if (tid < s) red[tid] = fmaxf(red[tid], red[tid+s]); __syncthreads(); }
    m = red[0]; __syncthreads();
    float sum = 0.f;
    #pragma unroll
    for (int j = 0; j < 8; ++j) { v[j] = __expf(v[j] - m); sum += v[j]; }
    red[tid] = sum; __syncthreads();
    for (int s = 128; s > 0; s >>= 1) { if (tid < s) red[tid] += red[tid+s]; __syncthreads(); }
    float inv = 1.f / red[0];
    #pragma unroll
    for (int j = 0; j < 8; ++j) row[tid + j*256] = v[j] * inv;
}

// act = bf16( sigmoid(g) * silu(u) )
__global__ void gateact_k(const float* __restrict__ g, const float* __restrict__ u,
                          u16* __restrict__ act)
{
    size_t i = (size_t)blockIdx.x * 256 + threadIdx.x;
    float gv = g[i], uv = u[i];
    float sg = 1.f / (1.f + __expf(-gv));
    float su = uv / (1.f + __expf(-uv));
    act[i] = f2bf(sg * su);
}

// Deterministic single-block reduction -> scalar
__global__ void reduce_k(const float* __restrict__ src, int n, float* __restrict__ dst,
                         float scale, int useabs)
{
    __shared__ float red[256];
    int tid = threadIdx.x; float s = 0.f;
    for (int i = tid; i < n; i += 256) { float v = src[i]; s += useabs ? fabsf(v) : v; }
    red[tid] = s; __syncthreads();
    for (int st = 128; st > 0; st >>= 1) { if (tid < st) red[tid] += red[tid+st]; __syncthreads(); }
    if (tid == 0) dst[0] = red[0] * scale;
}

// ---------------------------------------------------------------------------
// Host side
// ---------------------------------------------------------------------------
static inline void gemmNT(hipStream_t st, bool abf, bool obf,
                          const void* A, const u16* Bm, void* C,
                          int M, int N, int K, int lda, int ldb, int ldc,
                          long long sA1, long long sA2, long long sB1, long long sB2,
                          long long sC1, long long sC2, int zsplit, int batch, float alpha)
{
    dim3 g(N / 64, M / 128, batch), b(256);
    if (abf) {
        if (obf) gemm_nt_k<true , true ><<<g, b, 0, st>>>(A, Bm, C, K, lda, ldb, ldc,
                     sA1, sA2, sB1, sB2, sC1, sC2, zsplit, alpha);
        else     gemm_nt_k<true , false><<<g, b, 0, st>>>(A, Bm, C, K, lda, ldb, ldc,
                     sA1, sA2, sB1, sB2, sC1, sC2, zsplit, alpha);
    } else {
        if (obf) gemm_nt_k<false, true ><<<g, b, 0, st>>>(A, Bm, C, K, lda, ldb, ldc,
                     sA1, sA2, sB1, sB2, sC1, sC2, zsplit, alpha);
        else     gemm_nt_k<false, false><<<g, b, 0, st>>>(A, Bm, C, K, lda, ldb, ldc,
                     sA1, sA2, sB1, sB2, sC1, sC2, zsplit, alpha);
    }
}

static inline void cvtBF(hipStream_t st, const float* in, u16* out, size_t n)
{
    f32_to_bf16_k<<<(unsigned)(n / 1024), 256, 0, st>>>(in, out);
}

extern "C" void kernel_launch(void* const* d_in, const int* in_sizes, int n_in,
                              void* d_out, int out_size, void* d_ws, size_t ws_size,
                              hipStream_t stream)
{
    const float* x        = (const float*)d_in[0];
    const float* proto    = (const float*)d_in[2];
    const float* fam_w    = (const float*)d_in[3];
    const float* dict_w   = (const float*)d_in[4];
    const float* enc_w    = (const float*)d_in[5];
    const float* wq       = (const float*)d_in[6];
    const float* wk       = (const float*)d_in[7];
    const float* wv       = (const float*)d_in[8];
    const float* wo       = (const float*)d_in[9];
    const float* rel      = (const float*)d_in[10];
    const float* gate_w   = (const float*)d_in[11];
    const float* up_w     = (const float*)d_in[12];
    const float* down_w   = (const float*)d_in[13];
    const float* n1w      = (const float*)d_in[14];
    const float* n2w      = (const float*)d_in[15];
    const float* bias     = (const float*)d_in[16];

    // d_out layout: x | attn | family_scores | sparsity | e1 | e2
    float* out_x    = (float*)d_out;
    float* out_attn = out_x + (size_t)NTOK * DM;
    float* out_fam  = out_attn + (size_t)BSZ * NH * TSEQ * TSEQ;
    float* out_sp   = out_fam + (size_t)NTOK * NFAM;
    float* out_e1   = out_sp + 1;
    float* out_e2   = out_sp + 2;

    // ws layout (peak ~194 MB; aliases noted)
    char* ws = (char*)d_ws;
    const size_t MB = 1ull << 20;
    u16*   enc_wb   = (u16*)(ws + 0 * MB);          // 32MB
    u16*   wqb      = (u16*)(ws + 32 * MB);         // 2MB
    u16*   wkb      = (u16*)(ws + 34 * MB);
    u16*   wvb      = (u16*)(ws + 36 * MB);
    u16*   wob      = (u16*)(ws + 38 * MB);
    u16*   gateb    = (u16*)(ws + 40 * MB);         // 8MB
    u16*   upb      = (u16*)(ws + 48 * MB);         // 8MB
    u16*   downb    = (u16*)(ws + 56 * MB);         // 8MB
    u16*   famb     = (u16*)(ws + 64 * MB);         // 128KB
    u16*   relTb    = (u16*)(ws + 64 * MB + 131072);// 128KB
    float* tvb      = (float*)(ws + 64 * MB + 262144);        // 128KB
    int*   tib      = (int*)  (ws + 64 * MB + 262144 + 131072);
    float* rms1     = (float*)(ws + 64 * MB + 262144 + 262144);
    float* rms2     = (float*)(ws + 64 * MB + 262144 + 262144 + 16384);
    float* fam_raw  = (float*)(ws + 65 * MB);       // 1MB
    float* basis    = (float*)(ws + 66 * MB);       // 16MB (dead after sparse_norm)
    float* vlin     = (float*)(ws + 66 * MB);       //   alias basis (step 5+)
    float* resid    = (float*)(ws + 82 * MB);       // 16MB (dead after coeffs)
    float* tmp      = (float*)(ws + 82 * MB);       //   alias resid (step 10+)
    float* cbuf     = (float*)(ws + 98 * MB);       // 32MB (dead after topk)
    u16*   qlinb    = (u16*)(ws + 98 * MB);         //   alias cbuf[ 0: 8MB]
    u16*   klinb    = (u16*)(ws + 106 * MB);        //   alias cbuf[ 8:16MB]
    u16*   kmodb    = (u16*)(ws + 114 * MB);        //   alias cbuf[16:24MB]
    u16*   vTb      = (u16*)(ws + 122 * MB);        //   alias cbuf[24:32MB]
    u16*   normedb  = (u16*)(ws + 130 * MB);        // 8MB
    u16*   normed2b = (u16*)(ws + 138 * MB);        // 8MB
    u16*   oheadsb  = (u16*)(ws + 146 * MB);        // 8MB
    float* gbuf     = (float*)(ws + 154 * MB);      // 16MB
    float* ubuf     = (float*)(ws + 170 * MB);      // 16MB
    u16*   actb     = (u16*)(ws + 186 * MB);        // 8MB

    const long long Z = 0;

    // 0) weights -> bf16 (halves the dominant HBM streams; enables async DMA)
    cvtBF(stream, fam_w,  famb,  (size_t)NFAM * DM);
    cvtBF(stream, enc_w,  enc_wb, (size_t)NDICT * DM);
    cvtBF(stream, wq,     wqb,   (size_t)DM * DM);
    cvtBF(stream, wk,     wkb,   (size_t)DM * DM);
    cvtBF(stream, wv,     wvb,   (size_t)DM * DM);
    cvtBF(stream, wo,     wob,   (size_t)DM * DM);
    cvtBF(stream, gate_w, gateb, (size_t)DFF * DM);
    cvtBF(stream, up_w,   upb,   (size_t)DFF * DM);
    cvtBF(stream, down_w, downb, (size_t)DM * DFF);

    // 1) family gate scores + softmax -> out_fam
    gemmNT(stream, false, false, x, famb, fam_raw, NTOK, NFAM, DM, DM, DM, NFAM,
           Z,Z,Z,Z,Z,Z, 1, 1, 1.f);
    softmax64_k<<<NTOK / 8, 256, 0, stream>>>(fam_raw, out_fam);

    // 2) basis = fam @ proto ; resid = x - basis
    basis_k<<<NTOK, 256, 0, stream>>>(out_fam, proto, basis);
    sub_k<<<(NTOK * DM) / 256, 256, 0, stream>>>(x, basis, resid);

    // 3) coeffs = resid @ enc_w^T in 8 chunks of 512 tokens + top-8 select
    for (int c = 0; c < 8; ++c) {
        gemmNT(stream, false, false, resid + (size_t)c * 512 * DM, enc_wb, cbuf,
               512, NDICT, DM, DM, DM, NDICT, Z,Z,Z,Z,Z,Z, 1, 1, 1.f);
        topk8_k<<<512, 256, 0, stream>>>(cbuf, tvb, tib, c * 512);
    }

    // 4) fused sparse reconstruction + RMSNorm1 -> bf16 normed + rms1
    sparse_norm_k<<<NTOK, 256, 0, stream>>>(tvb, tib, basis, dict_w, bias, n1w,
                                            normedb, rms1);

    // 5) Q/K/V projections (A = bf16 normed; Q,K emitted bf16, V f32 for vT)
    gemmNT(stream, true, true , normedb, wqb, qlinb, NTOK, DM, DM, DM, DM, DM,
           Z,Z,Z,Z,Z,Z, 1, 1, 1.f);
    gemmNT(stream, true, true , normedb, wkb, klinb, NTOK, DM, DM, DM, DM, DM,
           Z,Z,Z,Z,Z,Z, 1, 1, 1.f);
    gemmNT(stream, true, false, normedb, wvb, vlin,  NTOK, DM, DM, DM, DM, DM,
           Z,Z,Z,Z,Z,Z, 1, 1, 1.f);

    // 6) k_mod[b,h] = K[b,h] @ rel[h]  (batched z = b*16+h), bf16 out
    transpose_rel_k<<<(NH * DH * DH) / 256, 256, 0, stream>>>(rel, relTb);
    gemmNT(stream, true, true, klinb, relTb, kmodb, TSEQ, DH, DH, DM, DH, DH,
           (long long)TSEQ * DM, DH,
           0, (long long)DH * DH,
           (long long)NH * TSEQ * DH, (long long)TSEQ * DH,
           NH, BSZ * NH, 1.f);

    // 7) V^T per (b,h) for the attn@V GEMM (bf16)
    transpose_vT_k<<<(NTOK * DM) / 256, 256, 0, stream>>>(vlin, vTb);

    // 8) scores = Q @ Kmod^T / 8 -> out_attn (f32), then row softmax in place
    gemmNT(stream, true, false, qlinb, kmodb, out_attn, TSEQ, TSEQ, DH, DM, DH, TSEQ,
           (long long)TSEQ * DM, DH,
           (long long)NH * TSEQ * DH, (long long)TSEQ * DH,
           (long long)NH * TSEQ * TSEQ, (long long)TSEQ * TSEQ,
           NH, BSZ * NH, 0.125f);
    attn_softmax_k<<<BSZ * NH * TSEQ, 256, 0, stream>>>(out_attn);

    // 9) out_heads[b,t,h*64+e] = attn @ V (A = f32 attn, converted in flight)
    gemmNT(stream, false, true, out_attn, vTb, oheadsb, TSEQ, DH, TSEQ, TSEQ, TSEQ, DM,
           (long long)NH * TSEQ * TSEQ, (long long)TSEQ * TSEQ,
           (long long)NH * DH * TSEQ, (long long)DH * TSEQ,
           (long long)TSEQ * DM, DH,
           NH, BSZ * NH, 1.f);

    // 10) attn output projection + residual -> out_x
    gemmNT(stream, true, false, oheadsb, wob, tmp, NTOK, DM, DM, DM, DM, DM,
           Z,Z,Z,Z,Z,Z, 1, 1, 1.f);
    add_k<<<(NTOK * DM) / 256, 256, 0, stream>>>(x, tmp, out_x);

    // 11) RMSNorm2 -> bf16 + rms2
    rmsnorm_k<<<NTOK, 256, 0, stream>>>(out_x, n2w, normed2b, rms2);

    // 12) gated FFN in 4 chunks of 1024 tokens, residual in place
    for (int c = 0; c < 4; ++c) {
        const u16* nchunk = normed2b + (size_t)c * 1024 * DM;
        gemmNT(stream, true, false, nchunk, gateb, gbuf, 1024, DFF, DM, DM, DM, DFF,
               Z,Z,Z,Z,Z,Z, 1, 1, 1.f);
        gemmNT(stream, true, false, nchunk, upb, ubuf, 1024, DFF, DM, DM, DM, DFF,
               Z,Z,Z,Z,Z,Z, 1, 1, 1.f);
        gateact_k<<<(1024 * DFF) / 256, 256, 0, stream>>>(gbuf, ubuf, actb);
        gemmNT(stream, true, false, actb, downb, tmp, 1024, DM, DFF, DFF, DFF, DM,
               Z,Z,Z,Z,Z,Z, 1, 1, 1.f);
        add_inplace_k<<<(1024 * DM) / 256, 256, 0, stream>>>(out_x + (size_t)c * 1024 * DM, tmp);
    }

    // 13) scalar outputs (deterministic single-block reductions)
    reduce_k<<<1, 256, 0, stream>>>(rms1, NTOK, out_e1, 1.f / NTOK, 0);
    reduce_k<<<1, 256, 0, stream>>>(rms2, NTOK, out_e2, 1.f / NTOK, 0);
    reduce_k<<<1, 256, 0, stream>>>(tvb, NTOK * TOPK, out_sp,
                                    1.f / ((float)NTOK * (float)NDICT), 1);
}
